// FreeFourierAttention_21698174779506
// MI455X (gfx1250) — matrix operand order repaired and checked by reading, not compiled, not run on hardware
//
#include <hip/hip_runtime.h>
#include <hip/hip_bf16.h>

typedef __attribute__((ext_vector_type(16))) _Float16 v16h;
typedef __attribute__((ext_vector_type(8)))  _Float16 v8h;
typedef __attribute__((ext_vector_type(8)))  float    v8f;

static constexpr int B_  = 4;
static constexpr int T_  = 2048;
static constexpr int C_  = 1024;
static constexpr int H_  = 16;
static constexpr int D_  = 64;
static constexpr int F_  = 256;     // bottleneck = C/4
static constexpr int BH_ = B_ * H_;
static constexpr int BT_ = B_ * T_;

#define PI_F    3.14159265358979323846f
#define RMSEPS  1.1920929e-07f

__device__ __forceinline__ float sigmoidf_(float x) { return 1.0f / (1.0f + expf(-x)); }

// Build a v16h WMMA fragment from two contiguous 16-byte (8-half) runs.
__device__ __forceinline__ v16h ld_frag16(const _Float16* p0, const _Float16* p1) {
  v8h lo = *(const v8h*)p0;
  v8h hi = *(const v8h*)p1;
  v16h r;
#pragma unroll
  for (int i = 0; i < 8; ++i) { r[i] = lo[i]; r[i + 8] = hi[i]; }
  return r;
}

// ---------------------------------------------------------------------------
// fp32 -> fp16 weight conversion
// ---------------------------------------------------------------------------
__global__ void f32_to_f16_kernel(const float* __restrict__ src,
                                  _Float16* __restrict__ dst, int n) {
  int i = blockIdx.x * blockDim.x + threadIdx.x;
  if (i < n) dst[i] = (_Float16)src[i];
}

// ---------------------------------------------------------------------------
// exp_w: per-row max, subtract, causal mask, exp, store f16 (zeros above diag)
// One block (256 threads) per row t.
// ---------------------------------------------------------------------------
__global__ __launch_bounds__(256) void prep_w_kernel(const float* __restrict__ w,
                                                     _Float16* __restrict__ ew) {
  const int t = blockIdx.x;
  const int tid = threadIdx.x;
  const float* row = w + (size_t)t * T_;

  float m = -3.4e38f;
  for (int j = tid; j < T_; j += 256) m = fmaxf(m, row[j]);
#pragma unroll
  for (int off = 16; off >= 1; off >>= 1) m = fmaxf(m, __shfl_xor(m, off, 32));

  __shared__ float smax[8];
  if ((tid & 31) == 0) smax[tid >> 5] = m;
  __syncthreads();
  if (tid == 0) {
    float mm = smax[0];
#pragma unroll
    for (int i = 1; i < 8; ++i) mm = fmaxf(mm, smax[i]);
    smax[0] = mm;
  }
  __syncthreads();
  m = smax[0];

  for (int j = tid; j < T_; j += 256) {
    float e = (j <= t) ? expf(row[j] - m) : 0.0f;
    ew[(size_t)t * T_ + j] = (_Float16)e;
  }
}

// ---------------------------------------------------------------------------
// QKV prep: sigmoid->fourier feats->3x3 mix->per-head RMS->rotary(q,k)->
// cross-head max(k)->exp_k. Writes sigmoid(q) [bh][t][d] (f16) and
// K-transposed PE tensor [bh][d'(128)][t] (f16): d'<64 => exp_k*v, d'>=64 => exp_k.
// One block (256 threads) per (b,t); 4 channels per thread (one head per thread).
// ---------------------------------------------------------------------------
__global__ __launch_bounds__(256) void prep_qkv_kernel(const float* __restrict__ x,
                                                       const float* __restrict__ kqv,
                                                       _Float16* __restrict__ sigq,
                                                       _Float16* __restrict__ pet) {
  __shared__ float sq[C_], sk[C_], sv[C_];
  const int bt  = blockIdx.x;
  const int b   = bt / T_;
  const int t   = bt % T_;
  const int tid = threadIdx.x;
  const int c0  = tid * 4;
  const int h   = c0 >> 6;

  const float k00 = kqv[0], k01 = kqv[1], k02 = kqv[2];
  const float k10 = kqv[3], k11 = kqv[4], k12 = kqv[5];
  const float k20 = kqv[6], k21 = kqv[7], k22 = kqv[8];

  float4 xv = *(const float4*)(x + (size_t)bt * C_ + c0);
  float xs[4] = {xv.x, xv.y, xv.z, xv.w};
  float q[4], k[4], v[4];
  float ssq = 0.f, ssk = 0.f, ssv = 0.f;
#pragma unroll
  for (int i = 0; i < 4; ++i) {
    float s  = sigmoidf_(xs[i]);
    float xn = 2.0f * PI_F * s - PI_F;
    float sn = sinf(xn), cs = cosf(xn), sc = sn * cs;
    q[i] = k00 * sn + k01 * cs + k02 * sc;
    k[i] = k10 * sn + k11 * cs + k12 * sc;
    v[i] = k20 * sn + k21 * cs + k22 * sc;
    ssq += q[i] * q[i]; ssk += k[i] * k[i]; ssv += v[i] * v[i];
  }
  // reduce sum-of-squares over the 16 lanes of this head (wave32)
#pragma unroll
  for (int off = 1; off < 16; off <<= 1) {
    ssq += __shfl_xor(ssq, off, 32);
    ssk += __shfl_xor(ssk, off, 32);
    ssv += __shfl_xor(ssv, off, 32);
  }
  float rq = rsqrtf(ssq * (1.0f / 64.0f) + RMSEPS);
  float rk = rsqrtf(ssk * (1.0f / 64.0f) + RMSEPS);
  float rv = rsqrtf(ssv * (1.0f / 64.0f) + RMSEPS);
#pragma unroll
  for (int i = 0; i < 4; ++i) {
    sq[c0 + i] = q[i] * rq;
    sk[c0 + i] = k[i] * rk;
    sv[c0 + i] = v[i] * rv;
  }
  __syncthreads();

  // Rotary. NB: the reference indexes cos/sin tables with the HEAD axis
  // (cos_tab[None,:H,None,:]), so theta = h * af[d2], independent of t.
  float qr[4], kr[4];
#pragma unroll
  for (int i = 0; i < 4; ++i) {
    int c  = c0 + i;
    int d  = c & 63;
    int d2 = (d < 32) ? d : (d - 32);
    float af = (d2 < 16) ? powf(1.0f / 1024.0f, (float)d2 * (1.0f / 15.0f)) : 0.0f;
    float th = (float)h * af;
    float cs = cosf(th), sn = sinf(th);
    if (d < 32) {
      qr[i] =  sq[c] * cs + sq[c + 32] * sn;
      kr[i] =  sk[c] * cs + sk[c + 32] * sn;
    } else {
      qr[i] = -sq[c - 32] * sn + sq[c] * cs;
      kr[i] = -sk[c - 32] * sn + sk[c] * cs;
    }
  }
  __syncthreads();
#pragma unroll
  for (int i = 0; i < 4; ++i) sk[c0 + i] = kr[i];   // rotary'd k back to LDS
  __syncthreads();

  const size_t bh = (size_t)(b * H_ + h);
#pragma unroll
  for (int i = 0; i < 4; ++i) {
    int c = c0 + i, d = c & 63;
    float mk = -3.4e38f;
#pragma unroll
    for (int hh = 0; hh < 16; ++hh) mk = fmaxf(mk, sk[hh * 64 + d]);
    float ek = expf(kr[i] - mk);
    size_t pbase = bh * (size_t)(128 * T_);
    pet[pbase + (size_t)d * T_ + t]        = (_Float16)(ek * sv[c]);   // P = exp_k * v
    pet[pbase + (size_t)(64 + d) * T_ + t] = (_Float16)ek;             // E = exp_k
    sigq[(bh * T_ + t) * D_ + d]           = (_Float16)sigmoidf_(qr[i]);
  }
}

// ---------------------------------------------------------------------------
// Attention matmul: per (b,h): [N|D] = exp_w(TxT, causal-zeroed) @ PE(Tx128).
// Block = 4 waves, 64 t-rows; wave = 16 rows x 128 cols = 8 f32 accumulators.
// Causal: K-loop only to the block's diagonal edge (exp_w is zero above diag).
// Epilogue: y = sigmoid(q) * n / d, stored f16 in (B,T,C).
// ---------------------------------------------------------------------------
__global__ __launch_bounds__(128) void attn_kernel(const _Float16* __restrict__ ew,
                                                   const _Float16* __restrict__ pet,
                                                   const _Float16* __restrict__ sigq,
                                                   _Float16* __restrict__ yh) {
  const int bh    = blockIdx.y;
  const int tb    = blockIdx.x;          // 64-row block index
  const int wave  = threadIdx.x >> 5;
  const int lane  = threadIdx.x & 31;
  const int cl    = lane & 15;
  const int g     = lane >> 4;
  const int tBase = tb * 64 + wave * 16;

  const _Float16* pe   = pet + (size_t)bh * 128 * T_;
  const _Float16* arow = ew + (size_t)(tBase + cl) * T_;

  v8f acc[8] = {};
  const int jEnd = tb * 64 + 64;
  for (int j = 0; j < jEnd; j += 32) {
    v16h a = ld_frag16(arow + j + 16 * g, arow + j + 16 * g + 8);
#pragma unroll
    for (int n = 0; n < 8; ++n) {
      const _Float16* bp = pe + (size_t)(n * 16 + cl) * T_ + j + 16 * g;
      v16h bb = ld_frag16(bp, bp + 8);
      acc[n] = __builtin_amdgcn_wmma_f32_16x16x32_f16(false, a, false, bb,
                                                      (short)0, acc[n], false, false);
    }
  }

  const int b = bh >> 4, h = bh & 15;
#pragma unroll
  for (int n = 0; n < 4; ++n) {
#pragma unroll
    for (int r = 0; r < 8; ++r) {
      int   t   = tBase + r + 8 * g;
      int   d   = n * 16 + cl;
      float num = acc[n][r];
      float den = acc[n + 4][r];
      float sg  = (float)sigq[((size_t)bh * T_ + t) * D_ + d];
      yh[((size_t)b * T_ + t) * C_ + h * 64 + d] = (_Float16)(sg * num / den);
    }
  }
}

// ---------------------------------------------------------------------------
// MLP1: h = relu(y @ w1^T)^2.  M=8192, N=256, K=1024.
// Block = 8 waves covering 64 rows x 256 cols; wave = 16 rows x 128 cols.
// ---------------------------------------------------------------------------
__global__ __launch_bounds__(256) void mlp1_kernel(const _Float16* __restrict__ yh,
                                                   const _Float16* __restrict__ w1h,
                                                   _Float16* __restrict__ hh) {
  const int wave  = threadIdx.x >> 5;
  const int lane  = threadIdx.x & 31;
  const int cl    = lane & 15;
  const int g     = lane >> 4;
  const int rBase = blockIdx.x * 64 + (wave >> 1) * 16;
  const int cBase = (wave & 1) * 128;

  const _Float16* arow = yh + (size_t)(rBase + cl) * C_;
  v8f acc[8] = {};
  for (int kk = 0; kk < C_; kk += 32) {
    v16h a = ld_frag16(arow + kk + 16 * g, arow + kk + 16 * g + 8);
#pragma unroll
    for (int n = 0; n < 8; ++n) {
      const _Float16* bp = w1h + (size_t)(cBase + n * 16 + cl) * C_ + kk + 16 * g;
      v16h bb = ld_frag16(bp, bp + 8);
      acc[n] = __builtin_amdgcn_wmma_f32_16x16x32_f16(false, a, false, bb,
                                                      (short)0, acc[n], false, false);
    }
  }
#pragma unroll
  for (int n = 0; n < 8; ++n) {
#pragma unroll
    for (int r = 0; r < 8; ++r) {
      float vr = fmaxf(acc[n][r], 0.0f);
      hh[(size_t)(rBase + r + 8 * g) * F_ + cBase + n * 16 + cl] = (_Float16)(vr * vr);
    }
  }
}

// ---------------------------------------------------------------------------
// MLP2: out = h @ w2^T.  M=8192, N=1024, K=256. Output fp32.
// ---------------------------------------------------------------------------
__global__ __launch_bounds__(256) void mlp2_kernel(const _Float16* __restrict__ hh,
                                                   const _Float16* __restrict__ w2h,
                                                   float* __restrict__ out) {
  const int wave  = threadIdx.x >> 5;
  const int lane  = threadIdx.x & 31;
  const int cl    = lane & 15;
  const int g     = lane >> 4;
  const int rBase = blockIdx.x * 64 + (wave >> 1) * 16;
  const int cBase = blockIdx.y * 256 + (wave & 1) * 128;

  const _Float16* arow = hh + (size_t)(rBase + cl) * F_;
  v8f acc[8] = {};
  for (int kk = 0; kk < F_; kk += 32) {
    v16h a = ld_frag16(arow + kk + 16 * g, arow + kk + 16 * g + 8);
#pragma unroll
    for (int n = 0; n < 8; ++n) {
      const _Float16* bp = w2h + (size_t)(cBase + n * 16 + cl) * F_ + kk + 16 * g;
      v16h bb = ld_frag16(bp, bp + 8);
      acc[n] = __builtin_amdgcn_wmma_f32_16x16x32_f16(false, a, false, bb,
                                                      (short)0, acc[n], false, false);
    }
  }
#pragma unroll
  for (int n = 0; n < 8; ++n) {
#pragma unroll
    for (int r = 0; r < 8; ++r) {
      out[(size_t)(rBase + r + 8 * g) * C_ + cBase + n * 16 + cl] = acc[n][r];
    }
  }
}

// ---------------------------------------------------------------------------
extern "C" void kernel_launch(void* const* d_in, const int* in_sizes, int n_in,
                              void* d_out, int out_size, void* d_ws, size_t ws_size,
                              hipStream_t stream) {
  const float* x   = (const float*)d_in[0];
  const float* kqv = (const float*)d_in[1];
  const float* w   = (const float*)d_in[2];
  const float* w1  = (const float*)d_in[3];
  const float* w2  = (const float*)d_in[4];
  float* out = (float*)d_out;
  (void)in_sizes; (void)n_in; (void)out_size; (void)ws_size;

  char*  ws  = (char*)d_ws;
  size_t off = 0;
  auto alloc = [&](size_t bytes) -> void* {
    void* p = ws + off;
    off += (bytes + 255) & ~(size_t)255;
    return p;
  };
  _Float16* ew   = (_Float16*)alloc((size_t)T_ * T_ * 2);          // 8 MB  exp_w (causal)
  _Float16* pet  = (_Float16*)alloc((size_t)BH_ * 128 * T_ * 2);   // 32 MB [bh][d'][t]
  _Float16* sigq = (_Float16*)alloc((size_t)BH_ * T_ * D_ * 2);    // 16 MB sigmoid(q)
  _Float16* yh   = (_Float16*)alloc((size_t)BT_ * C_ * 2);         // 16 MB y (B,T,C)
  _Float16* hh   = (_Float16*)alloc((size_t)BT_ * F_ * 2);         // 4 MB  hidden
  _Float16* w1h  = (_Float16*)alloc((size_t)F_ * C_ * 2);
  _Float16* w2h  = (_Float16*)alloc((size_t)C_ * F_ * 2);

  f32_to_f16_kernel<<<(F_ * C_ + 255) / 256, 256, 0, stream>>>(w1, w1h, F_ * C_);
  f32_to_f16_kernel<<<(C_ * F_ + 255) / 256, 256, 0, stream>>>(w2, w2h, C_ * F_);
  prep_w_kernel<<<T_, 256, 0, stream>>>(w, ew);
  prep_qkv_kernel<<<BT_, 256, 0, stream>>>(x, kqv, sigq, pet);
  attn_kernel<<<dim3(T_ / 64, BH_), 128, 0, stream>>>(ew, pet, sigq, yh);
  mlp1_kernel<<<BT_ / 64, 256, 0, stream>>>(yh, w1h, hh);
  mlp2_kernel<<<dim3(BT_ / 64, C_ / 256), 256, 0, stream>>>(hh, w2h, out);
}
